// RiverLSTMCellWithStatic_30081950941525
// MI455X (gfx1250) — matrix operand order repaired and checked by reading, not compiled, not run on hardware
//
#include <hip/hip_runtime.h>
#include <hip/hip_bf16.h>

typedef __attribute__((ext_vector_type(16))) _Float16 v16h;
typedef __attribute__((ext_vector_type(8)))  _Float16 v8h;
typedef __attribute__((ext_vector_type(8)))  float    v8f;

#define N_TOT   50000
#define KNB     8
#define HD      128
#define XD      32
#define SD      27
#define LIN     416    // 32 + 128 + 128 + 128
#define FOURH   512

// combined-buffer column offsets (f16 elements)
#define C_X   0
#define C_HS  32
#define C_NC  160
#define C_SP  288

// LDS strides (elements)
#define HIN_STR  136   // f16, padded
#define HS_STR   136
#define COMB_STR 424
#define Z_STR    520   // f32
#define HP_STR   132   // f32
#define ST_STR   28    // f32

// shared memory layout (byte offsets, all 16B aligned)
#define OFF_HSELF  0          // 16*136*2   = 4352
#define OFF_COMB   4352       // 16*424*2   = 13568
#define OFF_MASK   17920      // 128*4      = 512
#define OFF_STATIC 18432      // 16*28*4    = 1792
#define OFF_HPRE   20224      // 16*132*4   = 8448
#define OFF_OVL    28672      // max(128*136*2, 16*520*4) = 34816
#define SMEM_BYTES 63488

__device__ __forceinline__ float sigm(float x) { return 1.0f / (1.0f + __expf(-x)); }

// Build a 16-half fragment from two contiguous 16B pieces (LDS or global).
__device__ __forceinline__ v16h ld_frag(const _Float16* p0, const _Float16* p1) {
  v8h lo = *(const v8h*)p0;
  v8h hi = *(const v8h*)p1;
  return __builtin_shufflevector(lo, hi, 0,1,2,3,4,5,6,7,8,9,10,11,12,13,14,15);
}

__device__ __forceinline__ v8f wmma_f16(v16h a, v16h b, v8f c) {
  // (neg_a, A, neg_b, B, c_mod, C, reuse_a, reuse_b)
  return __builtin_amdgcn_wmma_f32_16x16x32_f16(false, a, false, b, (short)0, c, false, false);
}

// ---------------------------------------------------------------------------
// Prep: convert Wn / Wg / Wl to f16 into workspace (stays L2-resident: ~0.5MB)
// ---------------------------------------------------------------------------
__global__ void convert_weights_kernel(const float* __restrict__ Wn,
                                       const float* __restrict__ Wg,
                                       const float* __restrict__ Wl,
                                       _Float16* __restrict__ ws) {
  int i = blockIdx.x * blockDim.x + threadIdx.x;
  if (i < HD * HD)       ws[i] = (_Float16)Wn[i];
  if (i < HD * 2 * HD)   ws[HD * HD + i] = (_Float16)Wg[i];
  if (i < FOURH * LIN)   ws[HD * HD + HD * 2 * HD + i] = (_Float16)Wl[i];
}

// ---------------------------------------------------------------------------
// Fused river-LSTM cell: 16 samples per 256-thread workgroup (8 wave32s)
// ---------------------------------------------------------------------------
__global__ __launch_bounds__(256)
void river_cell_kernel(const float* __restrict__ x, const float* __restrict__ h_self,
                       const float* __restrict__ c_self, const float* __restrict__ h_inflows,
                       const int* __restrict__ inflow_valid, const float* __restrict__ statf,
                       const float* __restrict__ bg, const float* __restrict__ Ws,
                       const float* __restrict__ bs, const float* __restrict__ bl,
                       const float* __restrict__ ln_w, const float* __restrict__ ln_b,
                       const _Float16* __restrict__ wnH, const _Float16* __restrict__ wgH,
                       const _Float16* __restrict__ wlH,
                       float* __restrict__ h_out, float* __restrict__ c_out)
{
  __shared__ __align__(16) unsigned char smem[SMEM_BYTES];
  _Float16* sHself = (_Float16*)(smem + OFF_HSELF);
  _Float16* sComb  = (_Float16*)(smem + OFF_COMB);
  float*    sMask  = (float*)(smem + OFF_MASK);
  float*    sStat  = (float*)(smem + OFF_STATIC);
  float*    sHpre  = (float*)(smem + OFF_HPRE);
  _Float16* sHin   = (_Float16*)(smem + OFF_OVL);   // alive until barrier after pass 1
  float*    sZ     = (float*)(smem + OFF_OVL);      // alive after that barrier

  const int tid  = threadIdx.x;
  const int lane = tid & 31;
  const int wave = tid >> 5;
  const int l15  = lane & 15;
  const int hi16 = (lane >> 4) & 1;
  const int s0   = blockIdx.x * 16;     // base sample of this tile

  // ----------------- stage tiles into LDS (f32 -> f16) -----------------
  {
    // h_inflows: 128 rows (= 16 samples * 8 neighbors, contiguous) x 128
    const float4* src = (const float4*)(h_inflows + (size_t)s0 * KNB * HD);
    #pragma unroll
    for (int it = 0; it < 16; ++it) {
      int i4 = tid + it * 256;
      int row = i4 >> 5, c4 = i4 & 31;
      float4 v = src[i4];
      _Float16* d = sHin + row * HIN_STR + c4 * 4;
      d[0] = (_Float16)v.x; d[1] = (_Float16)v.y;
      d[2] = (_Float16)v.z; d[3] = (_Float16)v.w;
    }
  }
  {
    // h_self: 16x128 -> sHself and combined[:, 32:160]
    const float4* src = (const float4*)(h_self + (size_t)s0 * HD);
    #pragma unroll
    for (int it = 0; it < 2; ++it) {
      int i4 = tid + it * 256;
      int row = i4 >> 5, c4 = i4 & 31;
      float4 v = src[i4];
      _Float16* d1 = sHself + row * HS_STR + c4 * 4;
      _Float16* d2 = sComb + row * COMB_STR + C_HS + c4 * 4;
      d1[0] = d2[0] = (_Float16)v.x; d1[1] = d2[1] = (_Float16)v.y;
      d1[2] = d2[2] = (_Float16)v.z; d1[3] = d2[3] = (_Float16)v.w;
    }
  }
  if (tid < 128) {
    // x: 16x32 -> combined[:, 0:32]
    const float4* src = (const float4*)(x + (size_t)s0 * XD);
    int row = tid >> 3, c4 = tid & 7;
    float4 v = src[tid];
    _Float16* d = sComb + row * COMB_STR + C_X + c4 * 4;
    d[0] = (_Float16)v.x; d[1] = (_Float16)v.y;
    d[2] = (_Float16)v.z; d[3] = (_Float16)v.w;
  }
  if (tid < 128) sMask[tid] = (inflow_valid[(size_t)s0 * KNB + tid] != 0) ? 1.0f : 0.0f;
  for (int i = tid; i < 16 * SD; i += 256) {
    int row = i / SD, c = i - row * SD;
    sStat[row * ST_STR + c] = statf[(size_t)s0 * SD + i];
  }
  __syncthreads();

  // ----------------- static projection (K=27, fp32 VALU) -----------------
  {
    int h = tid & 127, sg = tid >> 7;
    float wrow[SD];
    #pragma unroll
    for (int j = 0; j < SD; ++j) wrow[j] = Ws[h * SD + j];
    float bsv = bs[h];
    #pragma unroll
    for (int ss = 0; ss < 8; ++ss) {
      int s = sg * 8 + ss;
      float acc = bsv;
      #pragma unroll
      for (int j = 0; j < SD; ++j) acc += sStat[s * ST_STR + j] * wrow[j];
      sComb[s * COMB_STR + C_SP + h] = (_Float16)acc;
    }
  }

  // ----------------- pass 1: gated neighbor context via WMMA -----------------
  {
    const int tileM   = wave;              // 16 (sample,k) rows per wave
    const int rowBase = tileM * 16;
    const int kA = hi16 ? 16 : 0;          // f16 A-layout K split
    const int kB = hi16 ? 16 : 0;          // f16 B-layout K split
    const _Float16* aRow  = sHin + (rowBase + l15) * HIN_STR;
    const _Float16* aRow2 = sHself + (tileM * 2 + (l15 >> 3)) * HS_STR;  // h_self half of gate_in
    for (int tN = 0; tN < 8; ++tN) {
      v8f mAcc = {};
      v8f gAcc = {};
      const _Float16* wnCol = wnH + (tN * 16 + l15) * HD;        // B col n = Wn row n
      const _Float16* wgCol = wgH + (tN * 16 + l15) * (2 * HD);  // B col n = Wg row n
      #pragma unroll
      for (int kb = 0; kb < 4; ++kb) {
        v16h a  = ld_frag(aRow + kb * 32 + kA, aRow + kb * 32 + kA + 8);
        v16h bm = ld_frag(wnCol + kb * 32 + kB, wnCol + kb * 32 + kB + 8);
        mAcc = wmma_f16(a, bm, mAcc);
        v16h b1 = ld_frag(wgCol + kb * 32 + kB, wgCol + kb * 32 + kB + 8);
        gAcc = wmma_f16(a, b1, gAcc);
      }
      #pragma unroll
      for (int kb = 0; kb < 4; ++kb) {      // gate_in cols 128..255 = h_self
        v16h a2 = ld_frag(aRow2 + kb * 32 + kA, aRow2 + kb * 32 + kA + 8);
        v16h b2 = ld_frag(wgCol + 128 + kb * 32 + kB, wgCol + 128 + kb * 32 + kB + 8);
        gAcc = wmma_f16(a2, b2, gAcc);
      }
      // epilogue: sigmoid gate * msg * mask, sum over this lane's 8 k-rows
      float bgv = bg[tN * 16 + l15];
      float ncv = 0.0f;
      const int off = hi16 * 8;             // C-layout: rows r+0 / r+8 per half-wave
      #pragma unroll
      for (int r = 0; r < 8; ++r) {
        float gate = sigm(gAcc[r] + bgv);
        ncv += gate * mAcc[r] * sMask[rowBase + off + r];
      }
      sComb[(tileM * 2 + hi16) * COMB_STR + C_NC + tN * 16 + l15] = (_Float16)ncv;
    }
  }
  __syncthreads();   // nc + sproj complete; sHin dead, sZ live from here on

  // ----------------- pass 2: z = combined @ Wl^T + bl via WMMA -----------------
  {
    const int kA = hi16 ? 16 : 0;
    const int kB = hi16 ? 16 : 0;
    const _Float16* aRow = sComb + l15 * COMB_STR;   // A row m = sample m
    #pragma unroll
    for (int t = 0; t < 4; ++t) {
      int tN = wave + t * 8;                         // 32 N-tiles over 512 outputs
      v8f acc = {};
      const _Float16* wlCol = wlH + (tN * 16 + l15) * LIN;
      #pragma unroll
      for (int kb = 0; kb < 13; ++kb) {
        v16h a = ld_frag(aRow + kb * 32 + kA, aRow + kb * 32 + kA + 8);
        v16h b = ld_frag(wlCol + kb * 32 + kB, wlCol + kb * 32 + kB + 8);
        acc = wmma_f16(a, b, acc);
      }
      float blv = bl[tN * 16 + l15];
      const int off = hi16 * 8;
      #pragma unroll
      for (int r = 0; r < 8; ++r)
        sZ[(r + off) * Z_STR + tN * 16 + l15] = acc[r] + blv;
    }
  }
  __syncthreads();

  // ----------------- LSTM elementwise (fp32) -----------------
  {
    int h = tid & 127, sg = tid >> 7;
    #pragma unroll
    for (int ss = 0; ss < 8; ++ss) {
      int s = sg * 8 + ss;
      float iv = sZ[s * Z_STR + h];
      float fv = sZ[s * Z_STR + 128 + h];
      float ov = sZ[s * Z_STR + 256 + h];
      float gv = sZ[s * Z_STR + 384 + h];
      float cs = c_self[(size_t)(s0 + s) * HD + h];
      float cn = sigm(fv) * cs + sigm(iv) * tanhf(gv);
      c_out[(size_t)(s0 + s) * HD + h] = cn;
      sHpre[s * HP_STR + h] = sigm(ov) * tanhf(cn);
    }
  }
  __syncthreads();

  // ----------------- LayerNorm over H=128; each wave32 owns 2 samples -----------------
  #pragma unroll
  for (int si = 0; si < 2; ++si) {
    int s = wave * 2 + si;
    float v0 = sHpre[s * HP_STR + lane];
    float v1 = sHpre[s * HP_STR + 32 + lane];
    float v2 = sHpre[s * HP_STR + 64 + lane];
    float v3 = sHpre[s * HP_STR + 96 + lane];
    float sum = v0 + v1 + v2 + v3;
    for (int d = 16; d >= 1; d >>= 1) sum += __shfl_xor(sum, d, 32);
    float mu = sum * (1.0f / 128.0f);
    float d0 = v0 - mu, d1 = v1 - mu, d2 = v2 - mu, d3 = v3 - mu;
    float sq = d0 * d0 + d1 * d1 + d2 * d2 + d3 * d3;
    for (int d = 16; d >= 1; d >>= 1) sq += __shfl_xor(sq, d, 32);
    float rs = rsqrtf(sq * (1.0f / 128.0f) + 1e-5f);
    float* hrow = h_out + (size_t)(s0 + s) * HD;
    hrow[lane]      = d0 * rs * ln_w[lane]      + ln_b[lane];
    hrow[32 + lane] = d1 * rs * ln_w[32 + lane] + ln_b[32 + lane];
    hrow[64 + lane] = d2 * rs * ln_w[64 + lane] + ln_b[64 + lane];
    hrow[96 + lane] = d3 * rs * ln_w[96 + lane] + ln_b[96 + lane];
  }
}

extern "C" void kernel_launch(void* const* d_in, const int* in_sizes, int n_in,
                              void* d_out, int out_size, void* d_ws, size_t ws_size,
                              hipStream_t stream) {
  const float* x         = (const float*)d_in[0];
  const float* h_self    = (const float*)d_in[1];
  const float* c_self    = (const float*)d_in[2];
  const float* h_inflows = (const float*)d_in[3];
  const int*   iv        = (const int*)d_in[4];
  const float* statf     = (const float*)d_in[5];
  const float* Wn        = (const float*)d_in[6];
  const float* Wg        = (const float*)d_in[7];
  const float* bg        = (const float*)d_in[8];
  const float* Ws        = (const float*)d_in[9];
  const float* bs        = (const float*)d_in[10];
  const float* Wl        = (const float*)d_in[11];
  const float* bl        = (const float*)d_in[12];
  const float* ln_w      = (const float*)d_in[13];
  const float* ln_b      = (const float*)d_in[14];

  _Float16* wsH = (_Float16*)d_ws;
  _Float16* wnH = wsH;                       // 128*128
  _Float16* wgH = wnH + HD * HD;             // 128*256
  _Float16* wlH = wgH + HD * 2 * HD;         // 512*416

  float* h_out = (float*)d_out;
  float* c_out = h_out + (size_t)N_TOT * HD;

  convert_weights_kernel<<<(FOURH * LIN + 255) / 256, 256, 0, stream>>>(Wn, Wg, Wl, wsH);
  river_cell_kernel<<<N_TOT / 16, 256, 0, stream>>>(
      x, h_self, c_self, h_inflows, iv, statf,
      bg, Ws, bs, bl, ln_w, ln_b, wnH, wgH, wlH, h_out, c_out);
}
